// DynamicsHead_82300163326344
// MI455X (gfx1250) — matrix operand order repaired and checked by reading, not compiled, not run on hardware
//
#include <hip/hip_runtime.h>

// ---------------------------------------------------------------------------
// CfC (Closed-form Continuous-time) 3-layer cell stack for MI455X (gfx1250).
// bf16 WMMA (V_WMMA_F32_16X16X32_BF16) with f32 accumulation; memory-bound.
//
// Layers (cat = in + hidden, K padded to mult of 32, N padded to mult of 16):
//   L0: cat=113 (Kpad 128), h=39 (Npad 48)
//   L1: cat= 64 (Kpad  64), h=25 (Npad 32)
//   L2: cat= 89 (Kpad  96), h=64 (Npad 64)
//
// CDNA5 paths used: v_wmma_f32_16x16x32_bf16 (LDS-fed b128 fragments),
// global_load_async_to_lds_b128 + s_wait_asynccnt for the weight block,
// global_prefetch for the next streaming row tile.
// ---------------------------------------------------------------------------

#define BATCH   131072
#define MTILES  (BATCH / 16)   // 8192 row tiles of 16
#define NBLK    256
#define WAVES   8

typedef __attribute__((ext_vector_type(16))) __bf16 v16bf;
typedef __attribute__((ext_vector_type(8)))  float  v8f;

union ABFrag { v16bf v; uint4 q[2]; };

__device__ __forceinline__ unsigned short f2bf(float f) {
  union { float f; unsigned int u; } cv; cv.f = f;
  unsigned int u = cv.u;
  return (unsigned short)((u + 0x7FFFu + ((u >> 16) & 1u)) >> 16);  // RNE
}

// One CfC layer on a 16-row tile held in LDS (row stride fixed at 128 bf16).
// sw: [4 gates][Npad][Kpad] bf16 (weights pre-transposed: row = N, col = K)
// out[nt*8 + r]: h_new value at (M = r + 8*(lane>>4), N = nt*16 + (lane&15))
template<int KSTEPS, int NTILES>
__device__ __forceinline__ void cfc_layer(
    const unsigned short* __restrict__ sx,
    const unsigned short* __restrict__ sw, int gstride, int Kpad,
    const float* __restrict__ sb, int bstride,
    int lane, float* out)
{
  const int m    = lane & 15;
  const int half = lane >> 4;

  // Preload all A fragments for this layer (contiguous K order, matching B):
  // element s : K = ks*32 + 16*half + s
  ABFrag a[KSTEPS];
#pragma unroll
  for (int ks = 0; ks < KSTEPS; ++ks) {
    const unsigned short* p = sx + m * 128 + ks * 32 + 16 * half;
    a[ks].q[0] = *(const uint4*)(p);        // K 0..7  (this half)
    a[ks].q[1] = *(const uint4*)(p + 8);    // K 8..15 (this half)
  }

#pragma unroll
  for (int nt = 0; nt < NTILES; ++nt) {
    v8f acc[4];
#pragma unroll
    for (int g = 0; g < 4; ++g) {           // ff1, ff2, time_a, time_b
      v8f c = {0.f, 0.f, 0.f, 0.f, 0.f, 0.f, 0.f, 0.f};
      // B layout: lane N = nt*16 + m, K base = ks*32 + 16*half, 16 consecutive K
      const unsigned short* wb = sw + g * gstride + (nt * 16 + m) * Kpad + half * 16;
#pragma unroll
      for (int ks = 0; ks < KSTEPS; ++ks) {
        ABFrag b;
        b.q[0] = *(const uint4*)(wb + ks * 32);
        b.q[1] = *(const uint4*)(wb + ks * 32 + 8);
        c = __builtin_amdgcn_wmma_f32_16x16x32_bf16(false, a[ks].v, false, b.v,
                                                    (short)0, c, false, false);
      }
      acc[g] = c;
    }
    const int n = nt * 16 + m;
    float b1 = sb[0 * bstride + n], b2 = sb[1 * bstride + n];
    float ba = sb[2 * bstride + n], bb = sb[3 * bstride + n];
#pragma unroll
    for (int r = 0; r < 8; ++r) {
      float f1 = tanhf(acc[0][r] + b1);
      float f2 = tanhf(acc[1][r] + b2);
      float t  = (acc[2][r] + ba) + (acc[3][r] + bb);   // ts = 1.0
      float s  = 1.0f / (1.0f + __expf(-t));            // sigmoid
      out[nt * 8 + r] = f1 * (1.0f - s) + s * f2;
    }
  }
}

__global__ __launch_bounds__(256) void cfc_main(
    const float* __restrict__ z,   const float* __restrict__ act,
    const float* __restrict__ clk, const float* __restrict__ hx,
    const unsigned short* __restrict__ gw, const float* __restrict__ gb,
    float* __restrict__ out_z, float* __restrict__ out_h)
{
  __shared__ __align__(16) unsigned short sW[57344];        // 114,688 B weights
  __shared__ float sB[576];                                 // padded biases
  __shared__ __align__(16) unsigned short sX[WAVES][16*128];// per-wave staging

  {  // async weight block copy: global -> LDS, bypassing VGPRs (ASYNCcnt)
    unsigned int lbase = (unsigned int)(uintptr_t)(&sW[0]);
    const char* gbase = (const char*)gw;
    for (int i = threadIdx.x; i < 57344 / 8; i += 256) {
      unsigned long long ga = (unsigned long long)(uintptr_t)(gbase + i * 16);
      unsigned int       la = lbase + i * 16;
      asm volatile("global_load_async_to_lds_b128 %0, %1, off"
                   :: "v"(la), "v"(ga) : "memory");
    }
    for (int i = threadIdx.x; i < 576; i += 256) sB[i] = gb[i];
    asm volatile("s_wait_asynccnt 0" ::: "memory");
  }
  __syncthreads();

  const int wave = threadIdx.x >> 5;
  const int lane = threadIdx.x & 31;
  const int m = lane & 15, half = lane >> 4;
  unsigned short* sx = sX[wave];

  for (int mt = blockIdx.x * WAVES + wave; mt < MTILES; mt += gridDim.x * WAVES) {
    const int row0 = mt * 16;

    // stage x_cat0 = [z(64) | action(8) | click(2) | h0(39) | pad] as bf16
    for (int i = lane; i < 16 * 128; i += 32) {
      int r = i >> 7, c = i & 127, gr = row0 + r;
      float v;
      if (c < 64)       v = z[(size_t)gr * 64 + c];
      else if (c < 72)  v = act[(size_t)gr * 8 + (c - 64)];
      else if (c < 74)  v = clk[(size_t)gr * 2 + (c - 72)];
      else if (c < 113) v = hx[(size_t)gr * 128 + (c - 74)];
      else              v = 0.0f;
      sx[i] = f2bf(v);
    }

    // prefetch next tile's streaming inputs while this tile computes
    {
      int mtn = mt + (int)gridDim.x * WAVES;
      if (mtn < MTILES) {
        const float* pz = z  + (size_t)mtn * 16 * 64;   //  4 KB tile
        const float* ph = hx + (size_t)mtn * 16 * 128;  //  8 KB tile
        __builtin_prefetch(pz + lane * 32, 0, 0);       // 32 x 128B lines
        __builtin_prefetch(ph + lane * 64, 0, 0);       // 32 x 256B spans
      }
    }

    // ---- layer 0: 113 -> 39 ----
    float h0[24];
    cfc_layer<4, 3>(sx, sW + 0, 6144, 128, sB + 0, 48, lane, h0);
#pragma unroll
    for (int nt = 0; nt < 3; ++nt) {
      int n = nt * 16 + m;
#pragma unroll
      for (int r = 0; r < 8; ++r) {
        if (n < 39) {
          int M = r + 8 * half;
          float v = h0[nt * 8 + r];
          out_h[(size_t)(row0 + M) * 128 + n] = v;
          sx[M * 128 + n] = f2bf(v);
        }
      }
    }
    for (int i = lane; i < 16 * 25; i += 32) {   // cols 39..63 <- hx[:,39:64]
      int r = i / 25, c = i % 25, gr = row0 + r;
      sx[r * 128 + 39 + c] = f2bf(hx[(size_t)gr * 128 + 39 + c]);
    }

    // ---- layer 1: 64 -> 25 ----
    float h1[16];
    cfc_layer<2, 2>(sx, sW + 24576, 2048, 64, sB + 192, 32, lane, h1);
#pragma unroll
    for (int nt = 0; nt < 2; ++nt) {
      int n = nt * 16 + m;
#pragma unroll
      for (int r = 0; r < 8; ++r) {
        if (n < 25) {
          int M = r + 8 * half;
          float v = h1[nt * 8 + r];
          out_h[(size_t)(row0 + M) * 128 + 39 + n] = v;
          sx[M * 128 + n] = f2bf(v);
        }
      }
    }
    for (int i = lane; i < 16 * 71; i += 32) {   // cols 25..88 <- hx[:,64:128]; 89..95 = 0
      int r = i / 71, c = 25 + i % 71, gr = row0 + r;
      float v = (c < 89) ? hx[(size_t)gr * 128 + 64 + (c - 25)] : 0.0f;
      sx[r * 128 + c] = f2bf(v);
    }

    // ---- layer 2: 89 -> 64 ----
    float h2[32];
    cfc_layer<3, 4>(sx, sW + 32768, 6144, 96, sB + 320, 64, lane, h2);
#pragma unroll
    for (int nt = 0; nt < 4; ++nt) {
      int n = nt * 16 + m;
#pragma unroll
      for (int r = 0; r < 8; ++r) {
        int M = r + 8 * half;
        float v = h2[nt * 8 + r];
        out_h[(size_t)(row0 + M) * 128 + 64 + n] = v;
        out_z[(size_t)(row0 + M) * 64 + n] = v;
      }
    }
  }
}

// ---------------------------------------------------------------------------
// Prep: build masked, padded, transposed bf16 weights + padded f32 biases in ws
// Workspace: [0, 114688) bytes: bf16 weights; then 576 f32 biases.
// ---------------------------------------------------------------------------
struct PrepArgs {
  const float* w[12];    // (l0:ff1,ff2,ta,tb)(l1:...)(l2:...), each (h, cat)
  const float* bia[12];  // matching biases, (h,)
  const float* msk[3];   // per-layer masks, (h, cat)
};

__global__ void cfc_prep(PrepArgs pa, unsigned short* wsw, float* wsb) {
  const int H_[3]   = {39, 25, 64};
  const int CAT_[3] = {113, 64, 89};
  const int KP[3]   = {128, 64, 96};
  const int NP[3]   = {48, 32, 64};
  const int WOFF[3] = {0, 24576, 32768};
  const int GST[3]  = {6144, 2048, 6144};
  const int BOFF[3] = {0, 192, 320};

  int l = blockIdx.x >> 2, g = blockIdx.x & 3;
  const float* W  = pa.w[l * 4 + g];
  const float* Bi = pa.bia[l * 4 + g];
  const float* Mk = pa.msk[l];
  int h = H_[l], cat = CAT_[l], Kp = KP[l], Np = NP[l];
  unsigned short* dst = wsw + WOFF[l] + g * GST[l];

  for (int i = threadIdx.x; i < Np * Kp; i += blockDim.x) {
    int n = i / Kp, k = i % Kp;
    float v = 0.0f;
    if (n < h && k < cat) {
      v = W[n * cat + k];
      if (g < 2) v *= Mk[n * cat + k];   // NCP mask only on ff1/ff2
    }
    dst[i] = f2bf(v);
  }
  for (int i = threadIdx.x; i < Np; i += blockDim.x)
    wsb[BOFF[l] + g * Np + i] = (i < h) ? Bi[i] : 0.0f;
}

// ---------------------------------------------------------------------------
extern "C" void kernel_launch(void* const* d_in, const int* in_sizes, int n_in,
                              void* d_out, int out_size, void* d_ws, size_t ws_size,
                              hipStream_t stream) {
  (void)in_sizes; (void)n_in; (void)out_size; (void)ws_size;

  const float* z   = (const float*)d_in[0];
  const float* act = (const float*)d_in[1];
  const float* clk = (const float*)d_in[2];
  const float* hx  = (const float*)d_in[3];

  PrepArgs pa;
  for (int l = 0; l < 3; ++l) {
    for (int g = 0; g < 4; ++g) {
      pa.w[l * 4 + g]   = (const float*)d_in[4 + l * 8 + g * 2];
      pa.bia[l * 4 + g] = (const float*)d_in[5 + l * 8 + g * 2];
    }
    pa.msk[l] = (const float*)d_in[28 + l];
  }

  unsigned short* wsw = (unsigned short*)d_ws;
  float* wsb = (float*)((char*)d_ws + 114688);

  cfc_prep<<<12, 256, 0, stream>>>(pa, wsw, wsb);

  float* out_z = (float*)d_out;
  float* out_h = out_z + (size_t)BATCH * 64;
  cfc_main<<<NBLK, 256, 0, stream>>>(z, act, clk, hx, wsw, wsb, out_z, out_h);
}
